// CharRNN_75496935129737
// MI455X (gfx1250) — compile-verified, matrix operand order checked by reading, not run on hardware
//
#include <hip/hip_runtime.h>
#include <hip/hip_bf16.h>

typedef __attribute__((ext_vector_type(16))) _Float16 v16h;
typedef __attribute__((ext_vector_type(8)))  _Float16 v8h;
typedef __attribute__((ext_vector_type(8)))  float    v8f;

#define B_   128
#define L_   512
#define V_   256
#define E_   64
#define H_   256

#define KTANH 2.885390081777927f   // 2*log2(e)

#define WMMA_F16(a, b, c) \
    __builtin_amdgcn_wmma_f32_16x16x32_f16(false, (a), false, (b), (short)0, (c), false, false)

// Branch-free tanh tail. gk is ALREADY pre-scaled by 2log2(e) (folded into
// xtab), so z = x*2log2(e) is one FMA.
// chain: v_fma -> v_exp_f32 -> v_add -> v_rcp_f32 -> v_fma. No EXEC divergence.
__device__ __forceinline__ float tanh_fused(float ccsum, float gk) {
    const float z = fmaf(ccsum, KTANH, gk);
#if __has_builtin(__builtin_amdgcn_exp2f)
    const float e = __builtin_amdgcn_exp2f(z);  // exp(2x)
#else
    const float e = exp2f(z);
#endif
    return 1.0f - 2.0f * __builtin_amdgcn_rcpf(e + 1.0f);
}

// Load one 16x32 f16 A-slice (ISA 7.12.2 layout) from an LDS h row.
__device__ __forceinline__ v16h load_a_slice(const _Float16* hrow, int s) {
    const v8h* ap = (const v8h*)(hrow + s * 32);
    return __builtin_shufflevector(ap[0], ap[2],
                                   0,1,2,3,4,5,6,7,8,9,10,11,12,13,14,15);
}

// ---------------------------------------------------------------------------
// Kernel 0: xtab[v][h] = (emb[v,:] @ W_xh[:,h] + b_h[h]) * 2log2(e)
// Embedding + input GEMM + bias + tanh pre-scale collapsed into one table.
// ---------------------------------------------------------------------------
__global__ void build_xtab_kernel(const float* __restrict__ emb,
                                  const float* __restrict__ W_xh,
                                  const float* __restrict__ b_h,
                                  float* __restrict__ xtab) {
    const int v = blockIdx.x;
    const int h = threadIdx.x;
    float acc = b_h[h];
#pragma unroll
    for (int e = 0; e < E_; ++e)
        acc = fmaf(emb[v * E_ + e], W_xh[e * H_ + h], acc);
    xtab[v * H_ + h] = acc * KTANH;
}

// ---------------------------------------------------------------------------
// Kernel 0b: pack W_hh / W_hy into f16 in the exact per-lane WMMA-B register
// order: flat index = ((ntile*8 + slice)*32 + lane)*16 + elem.
// The RNN kernel then loads Bh/By with contiguous 32-B vector loads, no cvt.
// ---------------------------------------------------------------------------
__global__ void pack_weights_kernel(const float* __restrict__ W_hh,
                                    const float* __restrict__ W_hy,
                                    _Float16* __restrict__ wBh,
                                    _Float16* __restrict__ wBy) {
    const int id   = blockIdx.x * 256 + threadIdx.x;  // 0 .. 65535
    const int i    = id & 15;                         // element in v16h
    const int lane = (id >> 4) & 31;
    const int s    = (id >> 9) & 7;
    const int nt   = (id >> 12) & 15;
    const int half = lane >> 4;
    const int l16  = lane & 15;
    const int ncol = nt * 16 + l16;
    const int kb   = s * 32 + half * 8;
    const int k    = (i < 8) ? (kb + i) : (kb + 16 + (i - 8));
    wBh[id] = (_Float16)W_hh[k * H_ + ncol];      // B[k][n] = W_hh[k][n]
    wBy[id] = (_Float16)W_hy[ncol * H_ + k];      // B[k][v] = W_hy[v][k]
}

// ---------------------------------------------------------------------------
// Kernel 1: fused recurrence + output head; iteration i runs BOTH GEMMs
// against the same A = h_{i-1}:
//     recurrence:  h_i          = tanh(h_{i-1} @ W_hh + xtab[x_i])   (cc0+cc1)
//     projection:  logits_{i-1} = h_{i-1} @ W_hy^T + bias            (o)
// ---------------------------------------------------------------------------
__global__ __launch_bounds__(512, 1)
void rnn_fused_kernel(const int*      __restrict__ x,       // [B, L]
                      const float*    __restrict__ hidden0, // [B, H]
                      const float*    __restrict__ xtab,    // [V, H] (pre-scaled)
                      const _Float16* __restrict__ wBh,     // packed f16 W_hh
                      const _Float16* __restrict__ wBy,     // packed f16 W_hy
                      const float*    __restrict__ W_hy_b,  // [V]
                      const float*    __restrict__ b_y,     // [V]
                      float* __restrict__ logits,           // [B, L, V]
                      float* __restrict__ final_h)          // [B, H]
{
    __shared__ int idx_s[16 * L_];                       // 32 KB tokens
    __shared__ __align__(16) _Float16 hbuf[2][16 * H_];  // 16 KB double-buffered h

    const int tid  = threadIdx.x;
    const int wave = tid >> 5;
    const int lane = tid & 31;
    const int half = lane >> 4;
    const int l16  = lane & 15;
    const int b0   = blockIdx.x * 16;
    const int ncol = wave * 16 + l16;

    for (int i = tid; i < 16 * L_; i += 512) {
        const int r = i >> 9, t = i & (L_ - 1);
        idx_s[i] = x[(b0 + r) * L_ + t];
    }
    for (int i = tid; i < 16 * H_; i += 512) {
        hbuf[0][i] = (_Float16)hidden0[(b0 + (i >> 8)) * H_ + (i & (H_ - 1))];
    }

    // Resident f16 B-tiles: straight 32-B vector loads, pre-packed layout.
    v16h Bh[8], By[8];
    {
        const v16h* pBh = (const v16h*)wBh + (wave * 8) * 32 + lane;
        const v16h* pBy = (const v16h*)wBy + (wave * 8) * 32 + lane;
#pragma unroll
        for (int s = 0; s < 8; ++s) Bh[s] = pBh[s * 32];
#pragma unroll
        for (int s = 0; s < 8; ++s) By[s] = pBy[s * 32];
    }
    const float biasv = W_hy_b[ncol] + b_y[ncol];

    __syncthreads();

    v8f c;  // current hidden state (f32, this wave's 16 columns)

    // ================= peeled i = 0 : recurrence only =================
    {
        float gk[8];
#pragma unroll
        for (int r = 0; r < 8; ++r)
            gk[r] = xtab[idx_s[(r + half * 8) * L_ + 0] * H_ + ncol];

        const _Float16* hrow = hbuf[0] + l16 * H_ + half * 8;
        v8f cc0 = {}, cc1 = {};
#pragma unroll
        for (int sg = 0; sg < 4; ++sg) {
            const v16h a0 = load_a_slice(hrow, 2 * sg + 0);
            const v16h a1 = load_a_slice(hrow, 2 * sg + 1);
            cc0 = WMMA_F16(a0, Bh[2 * sg + 0], cc0);
            cc1 = WMMA_F16(a1, Bh[2 * sg + 1], cc1);
        }
#pragma unroll
        for (int r = 0; r < 8; ++r) c[r] = tanh_fused(cc0[r] + cc1[r], gk[r]);
#pragma unroll
        for (int r = 0; r < 8; ++r)
            hbuf[1][(r + half * 8) * H_ + ncol] = (_Float16)c[r];
        __syncthreads();
    }

    // ============ main loop i = 1..511 : dual GEMM, shared A ============
#pragma unroll 1
    for (int i = 1; i < L_; ++i) {
        const _Float16* __restrict__ hp = hbuf[i & 1];        // h_{i-1}
        _Float16* __restrict__       hn = hbuf[(i + 1) & 1];  // h_i

        // gather issued first, consumed after all WMMAs (latency hidden)
        float gk[8];
#pragma unroll
        for (int r = 0; r < 8; ++r)
            gk[r] = xtab[idx_s[(r + half * 8) * L_ + i] * H_ + ncol];

        v8f cc0 = {}, cc1 = {};
        v8f o;
#pragma unroll
        for (int r = 0; r < 8; ++r) o[r] = biasv;

        const _Float16* hrow = hp + l16 * H_ + half * 8;
#pragma unroll
        for (int sg = 0; sg < 4; ++sg) {
            const v16h a0 = load_a_slice(hrow, 2 * sg + 0);
            const v16h a1 = load_a_slice(hrow, 2 * sg + 1);
            // interleave: no back-to-back RAW on any accumulator chain
            cc0 = WMMA_F16(a0, Bh[2 * sg + 0], cc0);
            o   = WMMA_F16(a0, By[2 * sg + 0], o);
            cc1 = WMMA_F16(a1, Bh[2 * sg + 1], cc1);
            o   = WMMA_F16(a1, By[2 * sg + 1], o);
        }

        // logits for t = i-1, non-temporal (write-once stream)
#pragma unroll
        for (int r = 0; r < 8; ++r)
            __builtin_nontemporal_store(
                o[r],
                &logits[((size_t)(b0 + r + half * 8) * L_ + (i - 1)) * V_ + ncol]);

#pragma unroll
        for (int r = 0; r < 8; ++r) c[r] = tanh_fused(cc0[r] + cc1[r], gk[r]);
#pragma unroll
        for (int r = 0; r < 8; ++r)
            hn[(r + half * 8) * H_ + ncol] = (_Float16)c[r];

        __syncthreads();
    }

    // ============ peeled i = 512 : projection of h_511 + epilogue ============
    {
        v8f o;
#pragma unroll
        for (int r = 0; r < 8; ++r) o[r] = biasv;

        const _Float16* hrow = hbuf[0] + l16 * H_ + half * 8;  // h_511
#pragma unroll
        for (int sg = 0; sg < 4; ++sg) {
            const v16h a0 = load_a_slice(hrow, 2 * sg + 0);
            const v16h a1 = load_a_slice(hrow, 2 * sg + 1);
            o = WMMA_F16(a0, By[2 * sg + 0], o);
            o = WMMA_F16(a1, By[2 * sg + 1], o);
        }
#pragma unroll
        for (int r = 0; r < 8; ++r)
            __builtin_nontemporal_store(
                o[r],
                &logits[((size_t)(b0 + r + half * 8) * L_ + (L_ - 1)) * V_ + ncol]);

        // final hidden state in full f32 from registers
#pragma unroll
        for (int r = 0; r < 8; ++r)
            final_h[(b0 + r + half * 8) * H_ + ncol] = c[r];
    }
}

// ---------------------------------------------------------------------------
extern "C" void kernel_launch(void* const* d_in, const int* in_sizes, int n_in,
                              void* d_out, int out_size, void* d_ws, size_t ws_size,
                              hipStream_t stream) {
    const int*   x      = (const int*)  d_in[0];
    const float* hidden = (const float*)d_in[1];
    const float* emb    = (const float*)d_in[2];
    const float* W_xh   = (const float*)d_in[3];
    const float* W_hh   = (const float*)d_in[4];
    const float* b_h    = (const float*)d_in[5];
    const float* W_hy_w = (const float*)d_in[6];
    const float* W_hy_b = (const float*)d_in[7];
    const float* b_y    = (const float*)d_in[8];

    float* logits  = (float*)d_out;
    float* final_h = logits + (size_t)B_ * L_ * V_;

    // workspace layout: xtab f32 [256*256] | wBh f16 [65536] | wBy f16 [65536]
    float*     xtab = (float*)d_ws;
    _Float16*  wBh  = (_Float16*)(xtab + V_ * H_);
    _Float16*  wBy  = wBh + 16 * 8 * 32 * 16;

    build_xtab_kernel<<<V_, H_, 0, stream>>>(emb, W_xh, b_h, xtab);
    pack_weights_kernel<<<256, 256, 0, stream>>>(W_hh, W_hy_w, wBh, wBy);
    rnn_fused_kernel<<<B_ / 16, 512, 0, stream>>>(
        x, hidden, xtab, wBh, wBy, W_hy_b, b_y, logits, final_h);
}
